// ResAdd_ws_dist_171798692037
// MI455X (gfx1250) — compile-verified, matrix operand order checked by reading, not run on hardware
//
#include <hip/hip_runtime.h>
#include <hip/hip_bf16.h>
#include <math.h>

typedef _Float16 v16h __attribute__((ext_vector_type(16)));
typedef float    v8f  __attribute__((ext_vector_type(8)));

// ---------------- problem constants ----------------
constexpr int B_  = 4;
constexpr int H_  = 28;
constexpr int W_  = 28;
constexpr int HW_ = H_ * W_;

constexpr int N_DS = 32768;   // 256*128
constexpr int N_C1 = 8192;    // 64*128
constexpr int N_C2 = 36864;   // 64*64*9
constexpr int N_C3 = 16384;   // 256*64

// workspace layout (float offsets)
constexpr int OFF_FDS  = 0;
constexpr int OFF_FC1  = OFF_FDS + N_DS;
constexpr int OFF_FC2  = OFF_FC1 + N_C1;
constexpr int OFF_FC3  = OFF_FC2 + N_C2;
constexpr int OFF_MM   = OFF_FC3 + N_C3;    // 8 floats: (min,max) x4
constexpr int OFF_KL   = OFF_MM  + 8;       // 4 floats
constexpr int OFF_STDS = OFF_KL  + 4;       // 256*2
constexpr int OFF_ST1  = OFF_STDS + 512;    // 64*2
constexpr int OFF_ST2  = OFF_ST1 + 128;     // 64*2
constexpr int OFF_ST3  = OFF_ST2 + 128;     // 256*2
constexpr int OFF_TDS  = OFF_ST3 + 512;     // 4*256*28*28 (adder-ds out -> residual in place)
constexpr int OFF_T1   = OFF_TDS + 802816;  // 4*64*28*28 (conv1 out)
constexpr int OFF_T2   = OFF_T1  + 200704;  // peg out -> h1 in place
constexpr int OFF_T3   = OFF_T2  + 200704;  // conv2 out -> h2 in place
constexpr int OFF_T4   = OFF_T3  + 200704;  // conv3 out

// Scalar |d| behind an optimizer-opaque barrier. The empty inline asm cannot
// be entered into an SLP vectorization tree and no combine can look through
// it, so the broken gfx1250 "v2f32 fabs -> i32 and/bitcast" lowering can
// never form. Scalar f32 fabs is a plain source modifier and always selects.
__device__ inline float fabs_opaque(float d) {
  float ad = __builtin_fabsf(d);
  asm("" : "+v"(ad));
  return ad;
}

// ---------------- block reduction helper ----------------
__device__ inline float block_reduce(float v, float* sm, int op) {
  int tid = threadIdx.x;
  sm[tid] = v;
  __syncthreads();
  for (int s = blockDim.x >> 1; s > 0; s >>= 1) {
    if (tid < s) {
      float a = sm[tid], b = sm[tid + s];
      sm[tid] = (op == 0) ? (a + b) : (op == 1 ? fmaxf(a, b) : fminf(a, b));
    }
    __syncthreads();
  }
  float r = sm[0];
  __syncthreads();
  return r;
}

// ---------------- min/max per weight tensor (4 blocks) ----------------
__global__ void minmax_kernel(const float* s0, const float* s1,
                              const float* s2, const float* s3, float* mm) {
  __shared__ float sm[1024];
  const float* p[4] = {s0, s1, s2, s3};
  const int ns[4] = {N_DS, N_C1, N_C2, N_C3};
  int t = blockIdx.x;
  const float* w = p[t];
  int n = ns[t];
  float mn = 3.4e38f, mx = -3.4e38f;
  for (int i = threadIdx.x; i < n; i += blockDim.x) {
    float v = w[i];
    mn = fminf(mn, v);
    mx = fmaxf(mx, v);
  }
  mn = block_reduce(mn, sm, 2);
  mx = block_reduce(mx, sm, 1);
  if (threadIdx.x == 0) { mm[2 * t] = mn; mm[2 * t + 1] = mx; }
}

// ---------------- bin reconstruction ----------------
__global__ void reconstruct_kernel(const float* __restrict__ w, const float* __restrict__ aff,
                                   const float* __restrict__ mm, int t,
                                   float* __restrict__ f, int n) {
  int i = blockIdx.x * blockDim.x + threadIdx.x;
  if (i >= n) return;
  float wmin = mm[2 * t], wmax = mm[2 * t + 1];
  float v = w[i];
  int idx = (int)floorf((v - wmin) / (wmax - wmin) * 100.0f);
  float mask = (idx < 100) ? 1.0f : 0.0f;
  int ci = idx < 0 ? 0 : (idx > 99 ? 99 : idx);
  f[i] = v * aff[ci] * mask;
}

// ---------------- KL(softmax(lap) || softmax(f)) per tensor (4 blocks) ------
__global__ void kl_kernel(const float* f0, const float* f1, const float* f2, const float* f3,
                          const float* l0, const float* l1, const float* l2, const float* l3,
                          float* klpart) {
  __shared__ float sm[1024];
  const float* fp[4] = {f0, f1, f2, f3};
  const float* lp[4] = {l0, l1, l2, l3};
  const int ns[4] = {N_DS, N_C1, N_C2, N_C3};
  int t = blockIdx.x;
  const float* f = fp[t];
  const float* lap = lp[t];
  int n = ns[t];
  float mf = -3.4e38f, ml = -3.4e38f;
  for (int i = threadIdx.x; i < n; i += blockDim.x) {
    mf = fmaxf(mf, f[i]);
    ml = fmaxf(ml, lap[i]);
  }
  mf = block_reduce(mf, sm, 1);
  ml = block_reduce(ml, sm, 1);
  float sf = 0.f, sl = 0.f;
  for (int i = threadIdx.x; i < n; i += blockDim.x) {
    sf += expf(f[i] - mf);
    sl += expf(lap[i] - ml);
  }
  sf = block_reduce(sf, sm, 0);
  sl = block_reduce(sl, sm, 0);
  float lseF = mf + logf(sf);
  float lseL = ml + logf(sl);
  float acc = 0.f;
  for (int i = threadIdx.x; i < n; i += blockDim.x) {
    float lpv = lap[i] - lseL;           // log p
    float p = expf(lpv);                 // p
    float lq = f[i] - lseF;              // log q
    acc += p * (lpv - lq);
  }
  acc = block_reduce(acc, sm, 0);
  if (threadIdx.x == 0) klpart[t] = acc / (float)n;
}

__global__ void finalize_kl_kernel(const float* klpart, float* dst) {
  if (threadIdx.x == 0 && blockIdx.x == 0)
    dst[0] = klpart[0] + klpart[1] + klpart[2] + klpart[3];
}

// ---------------- 1x1 adder conv via WMMA K-reduction ----------------
// out[b,o,y,x] = -sum_c |X[b,c,y,x] - W[o,c]|
// 2 waves/block; wave computes 16 x-positions x (OPB/2) output channels.
// A-fragment (16x32 f16): per-lane 16 abs-diffs; B = ones -> row sums into f32 C.
__global__ __launch_bounds__(64)
void adder1x1_wmma_kernel(const float* __restrict__ X, const float* __restrict__ Wt,
                          float* __restrict__ out, int B, int C, int H, int Wd,
                          int O, int OPB) {
  extern __shared__ float lds[];  // C * 16 floats
  const int ntx = (Wd + 15) >> 4;
  int tx = blockIdx.x % ntx;
  int by = blockIdx.x / ntx;
  int y = by % H, b = by / H;
  int x0 = tx << 4;
  int tid = threadIdx.x;
  for (int i = tid; i < C * 16; i += 64) {
    int c = i >> 4, mm = i & 15;
    int xg = x0 + mm;
    lds[i] = (xg < Wd) ? X[((b * C + c) * H + y) * Wd + xg] : 0.0f;
  }
  __syncthreads();
  int lane = tid & 31, wave = tid >> 5;
  int m = lane & 15, hf = lane >> 4;
  int nch = C >> 5;
  int opw = OPB >> 1;
  int obase = blockIdx.y * OPB + wave * opw;
  v16h ones;
#pragma unroll
  for (int e = 0; e < 16; ++e) ones[e] = (_Float16)1.0f;
  int r = lane & 7;
  bool wsel = (lane < 8) || (lane >= 16 && lane < 24);
  int row = r + (lane >= 16 ? 8 : 0);
  int xi = x0 + row;
  for (int oi = 0; oi < opw; ++oi) {
    int o = obase + oi;
    const float* wrow = Wt + o * C;
    __builtin_prefetch(wrow, 0, 0);
    v8f acc = {};
    for (int t = 0; t < nch; ++t) {
      v16h a;
#pragma unroll
      for (int e = 0; e < 16; ++e) {
        int c = t * 32 + ((e < 8) ? e : e + 8) + hf * 8;  // documented A layout
        a[e] = (_Float16)fabs_opaque(lds[(c << 4) + m] - wrow[c]);
      }
      acc = __builtin_amdgcn_wmma_f32_16x16x32_f16(false, a, false, ones,
                                                   (short)0, acc, false, false);
    }
    if (wsel && xi < Wd)
      out[((b * O + o) * H + y) * Wd + xi] = -acc[r];
  }
}

// ---------------- 3x3 (pad 1) adder conv, C=64, via WMMA K-reduction --------
__global__ __launch_bounds__(64)
void adder3x3_wmma_kernel(const float* __restrict__ X, const float* __restrict__ Wt,
                          float* __restrict__ out, int B, int H, int Wd,
                          int O, int OPB) {
  extern __shared__ float lds[];  // [64][3][18] = 3456 floats (zero padded halo)
  const int C = 64;
  const int ntx = (Wd + 15) >> 4;
  int tx = blockIdx.x % ntx;
  int by = blockIdx.x / ntx;
  int y = by % H, b = by / H;
  int x0 = tx << 4;
  int tid = threadIdx.x;
  for (int i = tid; i < C * 3 * 18; i += 64) {
    int c = i / 54, rj = i % 54;
    int dy = rj / 18, xx = rj % 18;
    int yi = y + dy - 1;
    int xg = x0 + xx - 1;
    float v = 0.0f;
    if (yi >= 0 && yi < H && xg >= 0 && xg < Wd)
      v = X[((b * C + c) * H + yi) * Wd + xg];
    lds[i] = v;
  }
  __syncthreads();
  int lane = tid & 31, wave = tid >> 5;
  int m = lane & 15, hf = lane >> 4;
  int opw = OPB >> 1;
  int obase = blockIdx.y * OPB + wave * opw;
  v16h ones;
#pragma unroll
  for (int e = 0; e < 16; ++e) ones[e] = (_Float16)1.0f;
  int r = lane & 7;
  bool wsel = (lane < 8) || (lane >= 16 && lane < 24);
  int row = r + (lane >= 16 ? 8 : 0);
  int xi = x0 + row;
  for (int oi = 0; oi < opw; ++oi) {
    int o = obase + oi;
    const float* wrow = Wt + o * 576;  // [c][ky][kx]
    __builtin_prefetch(wrow, 0, 0);
    v8f acc = {};
    for (int t = 0; t < 18; ++t) {      // 18 chunks of 32 K-values (2 c-halves x 9 taps)
      int kp = t % 9, ch = t / 9;
      int ky = kp / 3, kx = kp % 3;
      v16h a;
#pragma unroll
      for (int e = 0; e < 16; ++e) {
        int kk = ((e < 8) ? e : e + 8) + hf * 8;
        int c = ch * 32 + kk;
        float xv = lds[(c * 3 + ky) * 18 + m + kx];
        a[e] = (_Float16)fabs_opaque(xv - wrow[c * 9 + kp]);
      }
      acc = __builtin_amdgcn_wmma_f32_16x16x32_f16(false, a, false, ones,
                                                   (short)0, acc, false, false);
    }
    if (wsel && xi < Wd)
      out[((b * O + o) * H + y) * Wd + xi] = -acc[r];
  }
}

// ---------------- PEG depthwise 3x3 adder (pad 1) ----------------
__global__ void peg_dw_kernel(const float* __restrict__ X, const float* __restrict__ Wt,
                              float* __restrict__ out, int B, int C, int H, int Wd) {
  int i = blockIdx.x * blockDim.x + threadIdx.x;
  int n = B * C * H * Wd;
  if (i >= n) return;
  int x = i % Wd;
  int y = (i / Wd) % H;
  int c = (i / (Wd * H)) % C;
  int b = i / (Wd * H * C);
  const float* wr = Wt + c * 9;
  float s = 0.0f;
#pragma unroll
  for (int dy = 0; dy < 3; ++dy)
#pragma unroll
    for (int dx = 0; dx < 3; ++dx) {
      int yy = y + dy - 1, xx = x + dx - 1;
      float v = (yy >= 0 && yy < H && xx >= 0 && xx < Wd)
                    ? X[((b * C + c) * H + yy) * Wd + xx] : 0.0f;
      s += fabs_opaque(v - wr[dy * 3 + dx]);
    }
  out[i] = -s;
}

// ---------------- BatchNorm (training) stats: one block per channel --------
__global__ void bn_stats_kernel(const float* __restrict__ X, float* __restrict__ st,
                                int B, int C, int HW) {
  __shared__ float sm[256];
  int c = blockIdx.x;
  float s = 0.f, ss = 0.f;
  for (int i = threadIdx.x; i < B * HW; i += blockDim.x) {
    int b = i / HW, sp = i % HW;
    float v = X[(b * C + c) * HW + sp];
    s += v;
    ss += v * v;
  }
  s = block_reduce(s, sm, 0);
  ss = block_reduce(ss, sm, 0);
  if (threadIdx.x == 0) {
    float n = (float)(B * HW);
    float mean = s / n;
    float var = ss / n - mean * mean;
    var = var < 0.f ? 0.f : var;
    st[2 * c] = mean;
    st[2 * c + 1] = rsqrtf(var + 1e-5f);
  }
}

// ---------------- BN apply (+ optional residual, optional relu) ------------
__global__ void bn_apply_kernel(const float* __restrict__ X, const float* __restrict__ st,
                                const float* __restrict__ g, const float* __restrict__ bb,
                                const float* __restrict__ res, float* __restrict__ out,
                                int C, int HW, int n, int do_relu) {
  int i = blockIdx.x * blockDim.x + threadIdx.x;
  if (i >= n) return;
  int c = (i / HW) % C;
  float v = (X[i] - st[2 * c]) * st[2 * c + 1] * g[c] + bb[c];
  if (res) v += res[i];
  if (do_relu) v = fmaxf(v, 0.0f);
  out[i] = v;
}

// ---------------- host-side launch ----------------
extern "C" void kernel_launch(void* const* d_in, const int* in_sizes, int n_in,
                              void* d_out, int out_size, void* d_ws, size_t ws_size,
                              hipStream_t stream) {
  (void)in_sizes; (void)n_in; (void)ws_size;
  const float* x      = (const float*)d_in[0];
  const float* sw_ds  = (const float*)d_in[1];
  const float* sw_c1  = (const float*)d_in[2];
  const float* sw_c2  = (const float*)d_in[3];
  const float* sw_c3  = (const float*)d_in[4];
  const float* a_ds   = (const float*)d_in[5];
  const float* a_c1   = (const float*)d_in[6];
  const float* a_c2   = (const float*)d_in[7];
  const float* a_c3   = (const float*)d_in[8];
  const float* peg_w  = (const float*)d_in[9];
  const float* bn1_g  = (const float*)d_in[10];
  const float* bn1_b  = (const float*)d_in[11];
  const float* bn2_g  = (const float*)d_in[12];
  const float* bn2_b  = (const float*)d_in[13];
  const float* bn3_g  = (const float*)d_in[14];
  const float* bn3_b  = (const float*)d_in[15];
  const float* bnds_g = (const float*)d_in[16];
  const float* bnds_b = (const float*)d_in[17];
  const float* lap_ds = (const float*)d_in[18];
  const float* lap_c1 = (const float*)d_in[19];
  const float* lap_c2 = (const float*)d_in[20];
  const float* lap_c3 = (const float*)d_in[21];

  float* ws = (float*)d_ws;
  float* f_ds = ws + OFF_FDS;
  float* f_c1 = ws + OFF_FC1;
  float* f_c2 = ws + OFF_FC2;
  float* f_c3 = ws + OFF_FC3;
  float* mm   = ws + OFF_MM;
  float* klp  = ws + OFF_KL;
  float* stds = ws + OFF_STDS;
  float* st1  = ws + OFF_ST1;
  float* st2  = ws + OFF_ST2;
  float* st3  = ws + OFF_ST3;
  float* t_ds = ws + OFF_TDS;   // -> residual (in place)
  float* t1   = ws + OFF_T1;
  float* t2   = ws + OFF_T2;    // -> h1 (in place)
  float* t3   = ws + OFF_T3;    // -> h2 (in place)
  float* t4   = ws + OFF_T4;
  float* out  = (float*)d_out;

  // 1) weight min/max, bin reconstruction, KL terms
  minmax_kernel<<<4, 1024, 0, stream>>>(sw_ds, sw_c1, sw_c2, sw_c3, mm);
  reconstruct_kernel<<<(N_DS + 255) / 256, 256, 0, stream>>>(sw_ds, a_ds, mm, 0, f_ds, N_DS);
  reconstruct_kernel<<<(N_C1 + 255) / 256, 256, 0, stream>>>(sw_c1, a_c1, mm, 1, f_c1, N_C1);
  reconstruct_kernel<<<(N_C2 + 255) / 256, 256, 0, stream>>>(sw_c2, a_c2, mm, 2, f_c2, N_C2);
  reconstruct_kernel<<<(N_C3 + 255) / 256, 256, 0, stream>>>(sw_c3, a_c3, mm, 3, f_c3, N_C3);
  kl_kernel<<<4, 1024, 0, stream>>>(f_ds, f_c1, f_c2, f_c3,
                                    lap_ds, lap_c1, lap_c2, lap_c3, klp);

  const int ntx = (W_ + 15) / 16;                 // 2
  dim3 blk64(64);

  // 2) residual path: downsample adder (C=128 -> O=256) + BNds + relu (in place)
  adder1x1_wmma_kernel<<<dim3(B_ * H_ * ntx, 256 / 16), blk64, 128 * 16 * 4, stream>>>(
      x, f_ds, t_ds, B_, 128, H_, W_, 256, 16);
  bn_stats_kernel<<<256, 256, 0, stream>>>(t_ds, stds, B_, 256, HW_);
  bn_apply_kernel<<<(802816 + 255) / 256, 256, 0, stream>>>(
      t_ds, stds, bnds_g, bnds_b, nullptr, t_ds, 256, HW_, 802816, 1);

  // 3) main path: conv1 adder (C=128 -> O=64)
  adder1x1_wmma_kernel<<<dim3(B_ * H_ * ntx, 64 / 16), blk64, 128 * 16 * 4, stream>>>(
      x, f_c1, t1, B_, 128, H_, W_, 64, 16);
  // PEG depthwise 3x3 adder
  peg_dw_kernel<<<(200704 + 255) / 256, 256, 0, stream>>>(t1, peg_w, t2, B_, 64, H_, W_);
  // BN1 + relu (in place -> h1)
  bn_stats_kernel<<<64, 256, 0, stream>>>(t2, st1, B_, 64, HW_);
  bn_apply_kernel<<<(200704 + 255) / 256, 256, 0, stream>>>(
      t2, st1, bn1_g, bn1_b, nullptr, t2, 64, HW_, 200704, 1);

  // 4) conv2 adder 3x3 pad1 (C=64 -> O=64)
  adder3x3_wmma_kernel<<<dim3(B_ * H_ * ntx, 64 / 16), blk64, 64 * 3 * 18 * 4, stream>>>(
      t2, f_c2, t3, B_, H_, W_, 64, 16);
  bn_stats_kernel<<<64, 256, 0, stream>>>(t3, st2, B_, 64, HW_);
  bn_apply_kernel<<<(200704 + 255) / 256, 256, 0, stream>>>(
      t3, st2, bn2_g, bn2_b, nullptr, t3, 64, HW_, 200704, 1);

  // 5) conv3 adder (C=64 -> O=256), BN3, + residual, relu -> d_out
  adder1x1_wmma_kernel<<<dim3(B_ * H_ * ntx, 256 / 16), blk64, 64 * 16 * 4, stream>>>(
      t3, f_c3, t4, B_, 64, H_, W_, 256, 16);
  bn_stats_kernel<<<256, 256, 0, stream>>>(t4, st3, B_, 256, HW_);
  bn_apply_kernel<<<(802816 + 255) / 256, 256, 0, stream>>>(
      t4, st3, bn3_g, bn3_b, t_ds, out, 256, HW_, 802816, 1);

  // 6) KL scalar
  finalize_kl_kernel<<<1, 1, 0, stream>>>(klp, out + (out_size - 1));
}
